// Gaussian_55147380081240
// MI455X (gfx1250) — compile-verified, matrix operand order checked by reading, not run on hardware
//
#include <hip/hip_runtime.h>
#include <hip/hip_bf16.h>

typedef __bf16 bf16_t;
typedef __attribute__((ext_vector_type(8)))  __bf16 v8bf;
typedef __attribute__((ext_vector_type(16))) __bf16 v16bf;
typedef __attribute__((ext_vector_type(8)))  float  v8f;
typedef __attribute__((ext_vector_type(4)))  int    v4i;

#ifndef __has_builtin
#define __has_builtin(x) 0
#endif

#if __has_builtin(__builtin_amdgcn_global_load_async_to_lds_b128)
#define HAVE_ASYNC_LDS 1
typedef __attribute__((address_space(1))) v4i* gv4i_p;  // global int4*
typedef __attribute__((address_space(3))) v4i* sv4i_p;  // LDS int4*
#else
#define HAVE_ASYNC_LDS 0
#endif

__device__ __forceinline__ void wait_async0() {
#if __has_builtin(__builtin_amdgcn_s_wait_asynccnt)
    __builtin_amdgcn_s_wait_asynccnt(0);
#elif HAVE_ASYNC_LDS
    asm volatile("s_wait_asynccnt 0" ::: "memory");
#endif
}

__device__ __forceinline__ void nt_store(float v, float* p) {
#if __has_builtin(__builtin_nontemporal_store)
    __builtin_nontemporal_store(v, p);
#else
    *p = v;
#endif
}

namespace {
constexpr int N_  = 32768;
constexpr int K_  = 4096;
constexpr int NC_ = 4;
constexpr int D_  = 256;

constexpr int BM = 128;   // output-tile rows  (over N)
constexpr int BN = 128;   // output-tile cols  (over K)
constexpr int BK = 32;    // k-step, matches wmma_f32_16x16x32_bf16
constexpr int PITCH = 40; // LDS row pitch in bf16 (80 B; 64 B data + 16 B pad, conflict-free)
}

// One wave per row: emit bf16 copy (optionally pre-scaled by inv_std) and the
// inv_std-weighted squared row norm.
__global__ void prep_rows(const float* __restrict__ src, const float* __restrict__ stdv,
                          bf16_t* __restrict__ dstb, float* __restrict__ dsq,
                          long srcRowStride, int scaleOut) {
    const int wave = threadIdx.x >> 5;
    const int lane = threadIdx.x & 31;
    const long row = (long)blockIdx.x * 8 + wave;

    const float* r = src + row * srcRowStride + lane * 8;
    const float* s = stdv + lane * 8;

    float v[8], sd[8];
#pragma unroll
    for (int u = 0; u < 2; ++u) {
        float4 a = ((const float4*)r)[u];
        float4 b = ((const float4*)s)[u];
        v[u*4+0] = a.x; v[u*4+1] = a.y; v[u*4+2] = a.z; v[u*4+3] = a.w;
        sd[u*4+0] = b.x; sd[u*4+1] = b.y; sd[u*4+2] = b.z; sd[u*4+3] = b.w;
    }

    float part = 0.0f;
    v8bf pk;
#pragma unroll
    for (int e = 0; e < 8; ++e) {
        float inv = 1.0f / sd[e];
        part += v[e] * v[e] * inv;
        float o = scaleOut ? v[e] * inv : v[e];
        pk[e] = (bf16_t)o;
    }
    *(v8bf*)(dstb + row * D_ + lane * 8) = pk;

#pragma unroll
    for (int off = 16; off > 0; off >>= 1)
        part += __shfl_down(part, off, 32);
    if (lane == 0) dsq[row] = part;
}

// out[n,k] = exp(cross[n,k] - 0.5*(xsq[n] + musq[k]))
// cross = xb (N x D, bf16) * mus^T (K x D, bf16 pre-scaled by inv_std), f32 accum via WMMA.
__global__ __launch_bounds__(256)
void gemm_gauss(const bf16_t* __restrict__ xb, const bf16_t* __restrict__ mus,
                const float* __restrict__ xsq, const float* __restrict__ musq,
                float* __restrict__ out) {
    __shared__ bf16_t Als[2][BM * PITCH];
    __shared__ bf16_t Bls[2][BN * PITCH];

    const int tid  = threadIdx.x;
    const int lane = tid & 31;
    const int wave = tid >> 5;
    const int wm   = wave & 3;   // 0..3 : which 32-row band (over N)
    const int wn   = wave >> 2;  // 0..1 : which 64-col band (over K)
    const int lrow = lane & 15;
    const int half = lane >> 4;

    const int rowBase = blockIdx.y * BM;  // over N
    const int colBase = blockIdx.x * BN;  // over K

    v8f acc[2][4] = {};

    // Each thread moves 2 x 16B chunks per tile per matrix.
    const int c0row = tid >> 2;   // 0..63
    const int c0seg = tid & 3;    // 16B segment within a 64B row

    constexpr int KSTEPS = D_ / BK;  // 8

#if HAVE_ASYNC_LDS
    // --- Async DMA path: global -> LDS via GLOBAL_LOAD_ASYNC_TO_LDS_B128, double-buffered.
    auto tileLoad = [&](int t, int buf) {
#pragma unroll
        for (int u = 0; u < 2; ++u) {
            int rrow = c0row + u * 64;
            const bf16_t* ga = xb  + (size_t)(rowBase + rrow) * D_ + t * BK + c0seg * 8;
            const bf16_t* gb = mus + (size_t)(colBase + rrow) * D_ + t * BK + c0seg * 8;
            __builtin_amdgcn_global_load_async_to_lds_b128(
                (gv4i_p)(bf16_t*)ga, (sv4i_p)&Als[buf][rrow * PITCH + c0seg * 8], 0, 0);
            __builtin_amdgcn_global_load_async_to_lds_b128(
                (gv4i_p)(bf16_t*)gb, (sv4i_p)&Bls[buf][rrow * PITCH + c0seg * 8], 0, 0);
        }
    };

    tileLoad(0, 0);
    wait_async0();
    __syncthreads();

#pragma unroll 1
    for (int t = 0; t < KSTEPS; ++t) {
        const int cur = t & 1;
        if (t + 1 < KSTEPS) tileLoad(t + 1, cur ^ 1);

        v16bf a[2], b[4];
#pragma unroll
        for (int i = 0; i < 2; ++i) {
            int r = wm * 32 + i * 16 + lrow;
            v8bf lo = *(const v8bf*)(&Als[cur][r * PITCH + half * 8]);
            v8bf hi = *(const v8bf*)(&Als[cur][r * PITCH + 16 + half * 8]);
            a[i] = __builtin_shufflevector(lo, hi, 0,1,2,3,4,5,6,7,8,9,10,11,12,13,14,15);
        }
#pragma unroll
        for (int j = 0; j < 4; ++j) {
            int r = wn * 64 + j * 16 + lrow;
            v8bf lo = *(const v8bf*)(&Bls[cur][r * PITCH + half * 8]);
            v8bf hi = *(const v8bf*)(&Bls[cur][r * PITCH + 16 + half * 8]);
            b[j] = __builtin_shufflevector(lo, hi, 0,1,2,3,4,5,6,7,8,9,10,11,12,13,14,15);
        }

#pragma unroll
        for (int i = 0; i < 2; ++i)
#pragma unroll
            for (int j = 0; j < 4; ++j)
                acc[i][j] = __builtin_amdgcn_wmma_f32_16x16x32_bf16(
                    false, a[i], false, b[j], (short)0, acc[i][j], false, false);

        if (t + 1 < KSTEPS) {
            wait_async0();     // tile t+1 fully landed in LDS
            __syncthreads();   // all waves done reading buffer `cur`, t+1 visible to all
        }
    }
#else
    // --- Fallback: register-staged global -> LDS pipeline (round-1 path).
    auto gload = [&](const bf16_t* base, int rb, int kOff, uint4* rg) {
#pragma unroll
        for (int u = 0; u < 2; ++u) {
            int rrow = c0row + u * 64;
            rg[u] = *(const uint4*)(base + (size_t)(rb + rrow) * D_ + kOff + c0seg * 8);
        }
    };
    auto lstore = [&](const uint4* ra, const uint4* rb) {
#pragma unroll
        for (int u = 0; u < 2; ++u) {
            int rrow = c0row + u * 64;
            *(uint4*)(&Als[0][rrow * PITCH + c0seg * 8]) = ra[u];
            *(uint4*)(&Bls[0][rrow * PITCH + c0seg * 8]) = rb[u];
        }
    };

    uint4 aReg[2], bReg[2];
    gload(xb,  rowBase, 0, aReg);
    gload(mus, colBase, 0, bReg);
    lstore(aReg, bReg);
    __syncthreads();

#pragma unroll 1
    for (int t = 0; t < KSTEPS; ++t) {
        uint4 aNext[2], bNext[2];
        if (t + 1 < KSTEPS) {
            gload(xb,  rowBase, (t + 1) * BK, aNext);
            gload(mus, colBase, (t + 1) * BK, bNext);
        }

        v16bf a[2], b[4];
#pragma unroll
        for (int i = 0; i < 2; ++i) {
            int r = wm * 32 + i * 16 + lrow;
            v8bf lo = *(const v8bf*)(&Als[0][r * PITCH + half * 8]);
            v8bf hi = *(const v8bf*)(&Als[0][r * PITCH + 16 + half * 8]);
            a[i] = __builtin_shufflevector(lo, hi, 0,1,2,3,4,5,6,7,8,9,10,11,12,13,14,15);
        }
#pragma unroll
        for (int j = 0; j < 4; ++j) {
            int r = wn * 64 + j * 16 + lrow;
            v8bf lo = *(const v8bf*)(&Bls[0][r * PITCH + half * 8]);
            v8bf hi = *(const v8bf*)(&Bls[0][r * PITCH + 16 + half * 8]);
            b[j] = __builtin_shufflevector(lo, hi, 0,1,2,3,4,5,6,7,8,9,10,11,12,13,14,15);
        }

#pragma unroll
        for (int i = 0; i < 2; ++i)
#pragma unroll
            for (int j = 0; j < 4; ++j)
                acc[i][j] = __builtin_amdgcn_wmma_f32_16x16x32_bf16(
                    false, a[i], false, b[j], (short)0, acc[i][j], false, false);

        __syncthreads();
        if (t + 1 < KSTEPS) {
            lstore(aNext, bNext);
            __syncthreads();
        }
    }
#endif

    // Epilogue. C/D layout: VGPR v, lanes 0-15 -> M=v, lanes 16-31 -> M=v+8; N=lane%16.
    float xs[2][8];
#pragma unroll
    for (int i = 0; i < 2; ++i)
#pragma unroll
        for (int v = 0; v < 8; ++v)
            xs[i][v] = xsq[rowBase + wm * 32 + i * 16 + half * 8 + v];

#pragma unroll
    for (int j = 0; j < 4; ++j) {
        int gcol = colBase + wn * 64 + j * 16 + lrow;
        float mq = musq[gcol];
#pragma unroll
        for (int i = 0; i < 2; ++i) {
            int gr0 = rowBase + wm * 32 + i * 16 + half * 8;
#pragma unroll
            for (int v = 0; v < 8; ++v) {
                float lg = acc[i][j][v] - 0.5f * (xs[i][v] + mq);
                nt_store(__expf(lg), &out[(size_t)(gr0 + v) * K_ + gcol]);
            }
        }
    }
}

extern "C" void kernel_launch(void* const* d_in, const int* in_sizes, int n_in,
                              void* d_out, int out_size, void* d_ws, size_t ws_size,
                              hipStream_t stream) {
    const float* x    = (const float*)d_in[0];
    const float* mu   = (const float*)d_in[1];
    const float* stdv = (const float*)d_in[2];
    float* out = (float*)d_out;

    char* ws = (char*)d_ws;
    bf16_t* xb   = (bf16_t*)(ws);
    bf16_t* mus  = (bf16_t*)(ws + (size_t)N_ * D_ * sizeof(bf16_t));
    float*  xsq  = (float*) (ws + (size_t)(N_ + K_) * D_ * sizeof(bf16_t));
    float*  musq = (float*) (ws + (size_t)(N_ + K_) * D_ * sizeof(bf16_t) + (size_t)N_ * sizeof(float));

    prep_rows<<<N_ / 8, 256, 0, stream>>>(x,  stdv, xb,  xsq,  (long)D_,       0);
    prep_rows<<<K_ / 8, 256, 0, stream>>>(mu, stdv, mus, musq, (long)NC_ * D_, 1);
    gemm_gauss<<<dim3(K_ / BN, N_ / BM), 256, 0, stream>>>(xb, mus, xsq, musq, out);
}
